// CrossXMFusion_78245714198576
// MI455X (gfx1250) — matrix operand order repaired and checked by reading, compiled, not run on hardware
//
#include <hip/hip_runtime.h>
#include <hip/hip_bf16.h>
#include <math.h>

// Problem constants (fixed by the reference)
#define NB   16
#define CDIM 1024
#define TDIM 2048

typedef __bf16 v16bf __attribute__((ext_vector_type(16)));
typedef float  v8f   __attribute__((ext_vector_type(8)));
typedef unsigned int u32x4 __attribute__((ext_vector_type(4)));
typedef int          i32x4 __attribute__((ext_vector_type(4)));
typedef int          i32x8 __attribute__((ext_vector_type(8)));

union FragBF { v16bf v; uint4 q[2]; };
union AccU   { v8f v; float f[8]; };

#if defined(__AMDGCN__) && __has_builtin(__builtin_amdgcn_tensor_load_to_lds) && \
    __has_builtin(__builtin_amdgcn_s_wait_tensorcnt)
#define USE_TDM 1
#else
#define USE_TDM 0
#endif

__device__ __forceinline__ unsigned short f2bf(float f) {
    unsigned u = __builtin_bit_cast(unsigned, f);
    u += 0x7FFFu + ((u >> 16) & 1u);            // round-to-nearest-even
    return (unsigned short)(u >> 16);
}
__device__ __forceinline__ float bf2f(unsigned short h) {
    unsigned u = ((unsigned)h) << 16;
    return __builtin_bit_cast(float, u);
}
__device__ __forceinline__ float gelu_exact(float x) {
    return 0.5f * x * (1.0f + erff(x * 0.7071067811865475f));
}

// ---------------------------------------------------------------- converts
__global__ void k_convert_bf16(const float* __restrict__ in,
                               unsigned short* __restrict__ out, long n4) {
    long i = (long)blockIdx.x * blockDim.x + threadIdx.x;
    if (i >= n4) return;
    float4 v = *(const float4*)(in + i * 4);
    __attribute__((aligned(8))) unsigned short o[4] = {f2bf(v.x), f2bf(v.y), f2bf(v.z), f2bf(v.w)};
    *(ushort4*)(out + i * 4) = *(ushort4*)o;
}

// in: [NB][CDIM][TDIM] f32  ->  out: [NB][TDIM][CDIM] bf16
__global__ void k_transpose_convert(const float* __restrict__ in,
                                    unsigned short* __restrict__ out) {
    __shared__ float tile[32][33];
    int n = blockIdx.z;
    int t0 = blockIdx.x * 32, c0 = blockIdx.y * 32;
    const float* src = in + (size_t)n * CDIM * TDIM;
    unsigned short* dst = out + (size_t)n * CDIM * TDIM;
    int tx = threadIdx.x, ty = threadIdx.y;      // 32 x 8
#pragma unroll
    for (int j = 0; j < 32; j += 8)
        tile[ty + j][tx] = src[(size_t)(c0 + ty + j) * TDIM + t0 + tx];
    __syncthreads();
#pragma unroll
    for (int j = 0; j < 32; j += 8)
        dst[(size_t)(t0 + ty + j) * CDIM + c0 + tx] = f2bf(tile[tx][ty + j]);
}

// ---------------------------------------------------------------- softmax
// one 256-thread block per row of S[NB*CDIM][CDIM]; writes P (bf16) and P^T (bf16)
__global__ void k_softmax(const float* __restrict__ S,
                          unsigned short* __restrict__ P,
                          unsigned short* __restrict__ PT) {
    int row = blockIdx.x;                 // n*CDIM + c
    int n = row / CDIM, c = row % CDIM;
    int t = threadIdx.x;
    const float* s = S + (size_t)row * CDIM;
    float4 v = *(const float4*)(s + t * 4);

    __shared__ float red[16];
    float mx = fmaxf(fmaxf(v.x, v.y), fmaxf(v.z, v.w));
#pragma unroll
    for (int off = 16; off > 0; off >>= 1) mx = fmaxf(mx, __shfl_xor(mx, off));
    if ((t & 31) == 0) red[t >> 5] = mx;
    __syncthreads();
    mx = red[0];
#pragma unroll
    for (int i = 1; i < 8; i++) mx = fmaxf(mx, red[i]);

    float e0 = __expf(v.x - mx), e1 = __expf(v.y - mx);
    float e2 = __expf(v.z - mx), e3 = __expf(v.w - mx);
    float sm = e0 + e1 + e2 + e3;
#pragma unroll
    for (int off = 16; off > 0; off >>= 1) sm += __shfl_xor(sm, off);
    if ((t & 31) == 0) red[8 + (t >> 5)] = sm;
    __syncthreads();
    sm = red[8];
#pragma unroll
    for (int i = 1; i < 8; i++) sm += red[8 + i];
    float inv = 1.0f / sm;

    __attribute__((aligned(8))) unsigned short o[4] = {
        f2bf(e0 * inv), f2bf(e1 * inv), f2bf(e2 * inv), f2bf(e3 * inv)};
    *(ushort4*)(P + (size_t)row * CDIM + t * 4) = *(ushort4*)o;

    unsigned short* pt = PT + (size_t)n * CDIM * CDIM + c;
#pragma unroll
    for (int j = 0; j < 4; j++) pt[(size_t)(t * 4 + j) * CDIM] = o[j];
}

// ---------------------------------------------------------------- WMMA GEMM (NT)
// D[M,N] = sum_k A[m,k]*B[n,k]  with bf16 A,B; block tile 128x128, BK=32,
// 8 waves, each wave 64x32 (4x2 wmma tiles). Double-buffered LDS fed by the
// Tensor Data Mover (TDM) when available.
// MODE 0: out fp32 (normal layout), scaled by alpha                [stage 1]
// MODE 1: out bf16 transposed [N,M], + bf16 residual resT[n,m]     [stage 3]
// MODE 2: out bf16 transposed [N,M], + bias[m], exact GELU         [stage 4]
// MODE 3: out fp32 (normal), + bias[m] + fp32 residual (X|M)       [stage 5]
#define LDS_STRIDE 80           // 64B data + 16B pad  (bank-conflict-free)
#define TILE_BYTES (128 * LDS_STRIDE)

#if USE_TDM
// Issue a TDM DMA: 128 rows x 32 bf16 tile (row stride = ld elements) from
// global 'gptr' into LDS at byte offset 'lds_off', inserting 16B of padding
// after every 64B row so LDS rows land on an 80B stride.
__device__ __forceinline__ void tdm_load_tile(const unsigned short* gptr,
                                              unsigned lds_off, int ld) {
    unsigned long long ga = (unsigned long long)(uintptr_t)gptr;
    u32x4 g0;
    g0[0] = 1u;                                            // count=1 (valid), user mode
    g0[1] = lds_off;                                       // lds_addr
    g0[2] = (unsigned)(ga & 0xFFFFFFFFu);                  // global_addr[31:0]
    g0[3] = (unsigned)((ga >> 32) & 0x01FFFFFFu)           // global_addr[56:32]
            | (2u << 30);                                  // type = 2 ("image")
    const unsigned dim0 = (unsigned)ld;                    // tensor width (elements)
    const unsigned dim1 = 1u << 20;                        // plenty of rows (no OOB)
    const unsigned tile0 = 32, tile1 = 128;
    const unsigned long long s0 = (unsigned long long)ld;  // dim0 stride (elements)
    i32x8 g1;
    g1[0] = (int)((1u << 16)        // data_size = 2 bytes
                  | (1u << 20)      // pad_enable
                  | (3u << 22)      // pad_interval: 16 DW = 64B
                  | (3u << 25));    // pad_amount:   4 DW = 16B
    g1[1] = (int)((dim0 & 0xFFFFu) << 16);                         // dim0[15:0]
    g1[2] = (int)(((dim0 >> 16) & 0xFFFFu) | ((dim1 & 0xFFFFu) << 16));
    g1[3] = (int)(((dim1 >> 16) & 0xFFFFu) | (tile0 << 16));
    g1[4] = (int)(tile1 & 0xFFFFu);                                // tile_dim2 = 0
    g1[5] = (int)(unsigned)(s0 & 0xFFFFFFFFu);
    g1[6] = (int)(unsigned)((s0 >> 32) & 0xFFFFu);                 // dim1_stride = 0
    g1[7] = 0;
    i32x4 z4 = {};
#if __clang_major__ >= 23
    i32x8 z8 = {};
    __builtin_amdgcn_tensor_load_to_lds(g0, g1, z4, z4, z8, 0);
#else
    __builtin_amdgcn_tensor_load_to_lds(g0, g1, z4, z4, 0);
#endif
}
#endif  // USE_TDM

__device__ __forceinline__ void wmma_compute_step(const char* lds, int buf,
                                                  int waveM, int waveN, int lr,
                                                  int half, v8f (&acc)[4][2]) {
    const char* As = lds + buf * 2 * TILE_BYTES;
    const char* Bs = lds + (buf * 2 + 1) * TILE_BYTES;
    FragBF bfr[2];
#pragma unroll
    for (int jn = 0; jn < 2; jn++) {
        const char* p = Bs + (waveN * 32 + jn * 16 + lr) * LDS_STRIDE + half * 32;
        bfr[jn].q[0] = *(const uint4*)(p);
        bfr[jn].q[1] = *(const uint4*)(p + 16);
    }
#pragma unroll
    for (int im = 0; im < 4; im++) {
        FragBF af;
        const char* p = As + (waveM * 64 + im * 16 + lr) * LDS_STRIDE + half * 32;
        af.q[0] = *(const uint4*)(p);
        af.q[1] = *(const uint4*)(p + 16);
#pragma unroll
        for (int jn = 0; jn < 2; jn++)
            acc[im][jn] = __builtin_amdgcn_wmma_f32_16x16x32_bf16(
                false, af.v, false, bfr[jn].v, (short)0, acc[im][jn], false, false);
    }
}

template <int MODE>
__global__ __launch_bounds__(256)
void k_gemm_nt(const unsigned short* __restrict__ Abase,
               const unsigned short* __restrict__ Bbase,
               int lda, int ldb, int gk, float alpha,
               const float* __restrict__ bias,
               const unsigned short* __restrict__ resT, int ldrt,
               const float* __restrict__ resX, const float* __restrict__ resM,
               float* __restrict__ outF, int ldo,
               unsigned short* __restrict__ outT, int ldot,
               size_t strideA, size_t strideB, size_t strideO, size_t strideR) {
    __shared__ uint4 ldsq[4 * TILE_BYTES / 16];
    char* lds = (char*)ldsq;

    const int tid = threadIdx.x;
    const int bz = blockIdx.z;
    const unsigned short* A = Abase + strideA * bz;
    const unsigned short* B = Bbase + strideB * bz;
    const int mBase = blockIdx.x * 128;
    const int nBase = blockIdx.y * 128;

    const int wid = tid >> 5, lane = tid & 31;
    const int waveM = wid & 1;       // 0..1  (64 rows each)
    const int waveN = wid >> 1;      // 0..3  (32 cols each)
    const int lr = lane & 15, half = lane >> 4;

    v8f acc[4][2] = {};
    const int KT = gk / 32;

#if USE_TDM
    // -------- Tensor Data Mover pipeline: wave 0 drives the DMA engine.
    const unsigned ldsBase = (unsigned)(uintptr_t)lds;
    if (wid == 0) {
        tdm_load_tile(A + (size_t)mBase * lda, ldsBase + 0 * TILE_BYTES, lda);
        tdm_load_tile(B + (size_t)nBase * ldb, ldsBase + 1 * TILE_BYTES, ldb);
    }
    for (int kt = 0; kt < KT; ++kt) {
        const int buf = kt & 1;
        if (wid == 0) {
            if (kt + 1 < KT) {
                tdm_load_tile(A + (size_t)mBase * lda + (kt + 1) * 32,
                              ldsBase + ((buf ^ 1) * 2 + 0) * TILE_BYTES, lda);
                tdm_load_tile(B + (size_t)nBase * ldb + (kt + 1) * 32,
                              ldsBase + ((buf ^ 1) * 2 + 1) * TILE_BYTES, ldb);
                __builtin_amdgcn_s_wait_tensorcnt((short)2);  // tiles for 'kt' done
            } else {
                __builtin_amdgcn_s_wait_tensorcnt((short)0);
            }
        }
        __syncthreads();
        wmma_compute_step(lds, buf, waveM, waveN, lr, half, acc);
        __syncthreads();
    }
#else
    // -------- fallback: cooperative VGPR-staged loads, double-buffered LDS
    uint4 rga[2], rgb[2];
#define GLOAD(kt)                                                              \
    _Pragma("unroll") for (int cc = 0; cc < 2; cc++) {                         \
        int idx = tid + cc * 256;                                              \
        int row = idx >> 2, seg = idx & 3;                                     \
        rga[cc] = *(const uint4*)(A + (size_t)(mBase + row) * lda + (kt) * 32 + seg * 8); \
        rgb[cc] = *(const uint4*)(B + (size_t)(nBase + row) * ldb + (kt) * 32 + seg * 8); \
    }
#define STORE_LDS(buf)                                                         \
    _Pragma("unroll") for (int cc = 0; cc < 2; cc++) {                         \
        int idx = tid + cc * 256;                                              \
        int row = idx >> 2, seg = idx & 3;                                     \
        *(uint4*)(lds + (buf) * 2 * TILE_BYTES + row * LDS_STRIDE + seg * 16) = rga[cc]; \
        *(uint4*)(lds + ((buf) * 2 + 1) * TILE_BYTES + row * LDS_STRIDE + seg * 16) = rgb[cc]; \
    }
    GLOAD(0)
    STORE_LDS(0)
    __syncthreads();
    for (int kt = 0; kt < KT; ++kt) {
        int buf = kt & 1;
        if (kt + 1 < KT) { GLOAD(kt + 1) }
        wmma_compute_step(lds, buf, waveM, waveN, lr, half, acc);
        __syncthreads();
        if (kt + 1 < KT) {
            STORE_LDS(buf ^ 1)
            __syncthreads();
        }
    }
#undef GLOAD
#undef STORE_LDS
#endif

    // ---- epilogue
#pragma unroll
    for (int im = 0; im < 4; im++) {
#pragma unroll
        for (int jn = 0; jn < 2; jn++) {
            const int m0 = mBase + waveM * 64 + im * 16 + half * 8;  // 8 consecutive M rows
            const int n0 = nBase + waveN * 32 + jn * 16 + lr;        // single N column
            AccU a; a.v = acc[im][jn];
            if (MODE == 0) {
                float* o = outF + strideO * bz + (size_t)m0 * ldo + n0;
#pragma unroll
                for (int r = 0; r < 8; r++) o[(size_t)r * ldo] = a.f[r] * alpha;
            } else if (MODE == 1) {
                const unsigned short* rp = resT + strideR * bz + (size_t)n0 * ldrt + m0;
                ushort4 r0 = *(const ushort4*)(rp);
                ushort4 r1 = *(const ushort4*)(rp + 4);
                float rf[8] = {bf2f(r0.x), bf2f(r0.y), bf2f(r0.z), bf2f(r0.w),
                               bf2f(r1.x), bf2f(r1.y), bf2f(r1.z), bf2f(r1.w)};
                __attribute__((aligned(16))) unsigned short o8[8];
#pragma unroll
                for (int r = 0; r < 8; r++) o8[r] = f2bf(a.f[r] + rf[r]);
                unsigned short* dst = outT + strideO * bz + (size_t)n0 * ldot + m0;
                *(ushort4*)(dst) = *(ushort4*)(o8);
                *(ushort4*)(dst + 4) = *(ushort4*)(o8 + 4);
            } else if (MODE == 2) {
                __attribute__((aligned(16))) unsigned short o8[8];
#pragma unroll
                for (int r = 0; r < 8; r++)
                    o8[r] = f2bf(gelu_exact(a.f[r] + bias[m0 + r]));
                unsigned short* dst = outT + strideO * bz + (size_t)n0 * ldot + m0;
                *(ushort4*)(dst) = *(ushort4*)(o8);
                *(ushort4*)(dst + 4) = *(ushort4*)(o8 + 4);
            } else {  // MODE 3
                const float* rp = (bz < NB) ? resX + (size_t)bz * strideO
                                            : resM + (size_t)(bz - NB) * strideO;
                float* o = outF + strideO * bz + (size_t)m0 * ldo + n0;
#pragma unroll
                for (int r = 0; r < 8; r++)
                    o[(size_t)r * ldo] = a.f[r] + bias[m0 + r] + rp[(size_t)(m0 + r) * ldo + n0];
            }
        }
    }
}

// ---------------------------------------------------------------- launch
extern "C" void kernel_launch(void* const* d_in, const int* in_sizes, int n_in,
                              void* d_out, int out_size, void* d_ws, size_t ws_size,
                              hipStream_t stream) {
    (void)in_sizes; (void)n_in; (void)out_size; (void)ws_size;
    const float* X  = (const float*)d_in[0];
    const float* M  = (const float*)d_in[1];
    const float* W1 = (const float*)d_in[2];
    const float* b1 = (const float*)d_in[3];
    const float* W2 = (const float*)d_in[4];
    const float* b2 = (const float*)d_in[5];
    float* out = (float*)d_out;

    const size_t ECT = (size_t)NB * CDIM * TDIM;   // 33,554,432
    const size_t ECC = (size_t)NB * CDIM * CDIM;   // 16,777,216
    const size_t ECW = (size_t)CDIM * CDIM;

    unsigned short* Xb  = (unsigned short*)d_ws;     // [NB][C][T] bf16
    unsigned short* Mb  = Xb + ECT;                  // [NB][C][T]
    unsigned short* XT  = Mb + ECT;                  // [NB][T][C]
    unsigned short* MT  = XT + ECT;                  // [NB][T][C]
    unsigned short* W1b = MT + ECT;                  // [C][C]
    unsigned short* W2b = W1b + ECW;
    unsigned short* P   = W2b + ECW;                 // [NB][C][C]
    unsigned short* PT  = P + ECC;
    unsigned short* UT  = PT + ECC;                  // [2NB][T][C]
    float*          S   = (float*)(UT + 2 * ECT);    // [NB][C][C] fp32
    unsigned short* H1T = Xb;                        // reuse Xb+Mb (exactly 2*ECT)

    // 1) precision conversion + transposes
    k_convert_bf16<<<dim3((unsigned)(ECT / 4 / 256)), 256, 0, stream>>>(X, Xb, ECT / 4);
    k_convert_bf16<<<dim3((unsigned)(ECT / 4 / 256)), 256, 0, stream>>>(M, Mb, ECT / 4);
    k_convert_bf16<<<dim3((unsigned)(ECW / 4 / 256)), 256, 0, stream>>>(W1, W1b, ECW / 4);
    k_convert_bf16<<<dim3((unsigned)(ECW / 4 / 256)), 256, 0, stream>>>(W2, W2b, ECW / 4);
    k_transpose_convert<<<dim3(TDIM / 32, CDIM / 32, NB), dim3(32, 8), 0, stream>>>(X, XT);
    k_transpose_convert<<<dim3(TDIM / 32, CDIM / 32, NB), dim3(32, 8), 0, stream>>>(M, MT);

    // 2) S = (1/T) * Xb @ Mb^T   [C,C] fp32 per batch
    k_gemm_nt<0><<<dim3(CDIM / 128, CDIM / 128, NB), 256, 0, stream>>>(
        Xb, Mb, TDIM, TDIM, TDIM, 1.0f / (float)TDIM,
        nullptr, nullptr, 0, nullptr, nullptr,
        S, CDIM, nullptr, 0,
        (size_t)CDIM * TDIM, (size_t)CDIM * TDIM, (size_t)CDIM * CDIM, 0);

    // 3) row softmax -> P, P^T (bf16)
    k_softmax<<<dim3(NB * CDIM), 256, 0, stream>>>(S, P, PT);

    // 4a) U_x^T = (P^T @ XT^T + X)^T   stored [T,C] bf16 (batches 0..NB-1 of UT)
    k_gemm_nt<1><<<dim3(CDIM / 128, TDIM / 128, NB), 256, 0, stream>>>(
        PT, XT, CDIM, CDIM, CDIM, 1.0f,
        nullptr, XT, CDIM, nullptr, nullptr,
        nullptr, 0, UT, CDIM,
        (size_t)CDIM * CDIM, (size_t)CDIM * TDIM, (size_t)CDIM * TDIM, (size_t)CDIM * TDIM);

    // 4b) U_m^T = (P @ MT^T + M)^T     (batches NB..2NB-1 of UT)
    k_gemm_nt<1><<<dim3(CDIM / 128, TDIM / 128, NB), 256, 0, stream>>>(
        P, MT, CDIM, CDIM, CDIM, 1.0f,
        nullptr, MT, CDIM, nullptr, nullptr,
        nullptr, 0, UT + (size_t)NB * CDIM * TDIM, CDIM,
        (size_t)CDIM * CDIM, (size_t)CDIM * TDIM, (size_t)CDIM * TDIM, (size_t)CDIM * TDIM);

    // 5) H1^T = gelu(W1 @ U^T + b1)^T  [T,C] bf16, 2NB batches
    k_gemm_nt<2><<<dim3(CDIM / 128, TDIM / 128, 2 * NB), 256, 0, stream>>>(
        W1b, UT, CDIM, CDIM, CDIM, 1.0f,
        b1, nullptr, 0, nullptr, nullptr,
        nullptr, 0, H1T, CDIM,
        0, (size_t)CDIM * TDIM, (size_t)CDIM * TDIM, 0);

    // 6) out = W2 @ H1 + b2 + feature  [C,T] fp32 to d_out, 2NB batches
    k_gemm_nt<3><<<dim3(CDIM / 128, TDIM / 128, 2 * NB), 256, 0, stream>>>(
        W2b, H1T, CDIM, CDIM, CDIM, 1.0f,
        b2, nullptr, 0, X, M,
        out, TDIM, nullptr, 0,
        0, (size_t)CDIM * TDIM, (size_t)CDIM * TDIM, 0);
}
